// MoEMLP_36481452212931
// MI455X (gfx1250) — matrix operand order repaired and checked by reading, not compiled, not run on hardware
//
#include <hip/hip_runtime.h>
#include <hip/hip_bf16.h>
#include <stdint.h>

// ---- problem constants (from reference setup_inputs) ----
#define T_TOK 4096   // B*S = 2*2048
#define DDIM  1024
#define FDIM  2048
#define NEXP  8
#define TOPK  2
#define LDF   40     // padded K-stride (ushorts) for 32-wide K tiles: 80B rows

typedef __attribute__((ext_vector_type(16))) __bf16 v16bf;
typedef __attribute__((ext_vector_type(8)))  float  v8f;

union FragAB { v16bf v; uint4 q[2]; };

static __device__ __forceinline__ unsigned short f2bf(float f) {
    unsigned u = __float_as_uint(f);
    unsigned r = (u + 0x7FFFu + ((u >> 16) & 1u)) >> 16;
    return (unsigned short)r;
}
static __device__ __forceinline__ unsigned pk2bf(float lo, float hi) {
    return (unsigned)f2bf(lo) | ((unsigned)f2bf(hi) << 16);
}

// ---------------- kernel 1: convert x to bf16 ----------------
__global__ __launch_bounds__(256) void k_cvt_x(const float* __restrict__ x,
                                               unsigned short* __restrict__ xb) {
    size_t i = ((size_t)blockIdx.x * 256 + threadIdx.x) * 4;
    float4 v = *(const float4*)(x + i);
    uint2 o;
    o.x = pk2bf(v.x, v.y);
    o.y = pk2bf(v.z, v.w);
    *(uint2*)(xb + i) = o;
}

// ---------------- kernel 2: router (one wave32 per token) ----------------
__global__ __launch_bounds__(256) void k_router(const float* __restrict__ x,
                                                const float* __restrict__ gw,
                                                float* __restrict__ logits_out,
                                                int* __restrict__ sel,
                                                float* __restrict__ wsel) {
    const int wave = threadIdx.x >> 5;
    const int lane = threadIdx.x & 31;
    const int t = blockIdx.x * 8 + wave;

    float acc[NEXP];
#pragma unroll
    for (int e = 0; e < NEXP; ++e) acc[e] = 0.f;

    const float* xp = x + (size_t)t * DDIM;
    for (int d = lane * 4; d < DDIM; d += 32 * 4) {
        float4 xv = *(const float4*)(xp + d);
#pragma unroll
        for (int e = 0; e < NEXP; ++e) {
            float4 gv = *(const float4*)(gw + (size_t)e * DDIM + d);
            acc[e] += xv.x * gv.x + xv.y * gv.y + xv.z * gv.z + xv.w * gv.w;
        }
    }
#pragma unroll
    for (int e = 0; e < NEXP; ++e) {
#pragma unroll
        for (int off = 16; off > 0; off >>= 1)
            acc[e] += __shfl_xor(acc[e], off, 32);
    }
    if (lane == 0) {
#pragma unroll
        for (int e = 0; e < NEXP; ++e) logits_out[(size_t)t * NEXP + e] = acc[e];
        // top-2 on composite = -logit + e*1e-6 (ascending), matches reference argsort
        float b0 = 1e30f, b1 = 1e30f; int e0 = 0, e1 = 0;
#pragma unroll
        for (int e = 0; e < NEXP; ++e) {
            float c = -acc[e] + (float)e * 1e-6f;
            if (c < b0)      { b1 = b0; e1 = e0; b0 = c; e0 = e; }
            else if (c < b1) { b1 = c;  e1 = e; }
        }
        float l0 = acc[e0], l1 = acc[e1];
        float m  = fmaxf(l0, l1);
        float w0 = __expf(l0 - m), w1 = __expf(l1 - m);
        float inv = 1.f / (w0 + w1);
        sel[t * 2 + 0] = e0;  sel[t * 2 + 1] = e1;
        wsel[t * 2 + 0] = w0 * inv;  wsel[t * 2 + 1] = w1 * inv;
    }
}

// ---------------- kernels 3/4: bucket slots per expert ----------------
__global__ void k_zero_cnt(int* cnt) { if (threadIdx.x < NEXP) cnt[threadIdx.x] = 0; }

__global__ __launch_bounds__(256) void k_scatter(const int* __restrict__ sel,
                                                 int* __restrict__ cnt,
                                                 int* __restrict__ tix) {
    int i = blockIdx.x * 256 + threadIdx.x;   // slot id p = t*2+k
    if (i < T_TOK * TOPK) {
        int e = sel[i];
        int s = atomicAdd(&cnt[e], 1);
        tix[e * T_TOK + s] = i;
    }
}

// ---------------- kernel 5: gate+up GEMM + SiLU (bf16 WMMA) ----------------
// M-tile 128 (8 waves x 16 rows), N-tile 64 (4 sub-tiles per wave), K-step 32.
__global__ __launch_bounds__(256) void k_gemm1(const unsigned short* __restrict__ xb,
                                               const float* __restrict__ Wg,
                                               const float* __restrict__ Wu,
                                               const int* __restrict__ tix,
                                               const int* __restrict__ cnt,
                                               unsigned short* __restrict__ hbuf) {
    const int e = blockIdx.z;
    const int mtile = blockIdx.y;
    const int f0 = blockIdx.x * 64;
    const int ce = cnt[e];
    if (mtile * 128 >= ce) return;

    __shared__ int toks[128];
    __shared__ unsigned short lA[128 * LDF];
    __shared__ unsigned short lBg[64 * LDF];
    __shared__ unsigned short lBu[64 * LDF];

    const int tid = threadIdx.x;
    if (tid < 128) {
        int idx = mtile * 128 + tid;
        toks[tid] = (idx < ce) ? tix[e * T_TOK + idx] : 0;
    }
    __syncthreads();

    const int wave = tid >> 5, lane = tid & 31;
    const int wm = wave;                  // 8 wave-rows of 16
    const int half = lane >> 4, lr = lane & 15;

    // A staging: 512 chunk-assignments (128 rows x 4 x 16B), 2 per thread
    const int ar0 = tid >> 2, ac4 = tid & 3;
    const size_t abase0 = (size_t)(toks[ar0] >> 1) * DDIM + (size_t)ac4 * 8;
    const size_t abase1 = (size_t)(toks[ar0 + 64] >> 1) * DDIM + (size_t)ac4 * 8;

    // B staging: 256 assignments: d-pair (16) x f-group (16)
    const int dp = tid >> 4;              // d = 2*dp, 2*dp+1
    const int fg = (tid & 15) * 4;        // f = fg..fg+3

    const float* wgp = Wg + (size_t)e * DDIM * FDIM + f0;
    const float* wup = Wu + (size_t)e * DDIM * FDIM + f0;

    v8f zero = {0.f, 0.f, 0.f, 0.f, 0.f, 0.f, 0.f, 0.f};
    v8f cg[4], cu[4];
#pragma unroll
    for (int ns = 0; ns < 4; ++ns) { cg[ns] = zero; cu[ns] = zero; }

    for (int k0 = 0; k0 < DDIM; k0 += 32) {
        // stage A: [m][k]
        *(uint4*)&lA[ar0 * LDF + ac4 * 8]        = *(const uint4*)(xb + abase0 + k0);
        *(uint4*)&lA[(ar0 + 64) * LDF + ac4 * 8] = *(const uint4*)(xb + abase1 + k0);
        // stage B transposed [n=f][k=d], packed pairs of d -> b32 stores
        {
            const float4 g0 = *(const float4*)(wgp + (size_t)(k0 + 2 * dp + 0) * FDIM + fg);
            const float4 g1 = *(const float4*)(wgp + (size_t)(k0 + 2 * dp + 1) * FDIM + fg);
            const float4 u0 = *(const float4*)(wup + (size_t)(k0 + 2 * dp + 0) * FDIM + fg);
            const float4 u1 = *(const float4*)(wup + (size_t)(k0 + 2 * dp + 1) * FDIM + fg);
            *(unsigned*)&lBg[(fg + 0) * LDF + 2 * dp] = pk2bf(g0.x, g1.x);
            *(unsigned*)&lBg[(fg + 1) * LDF + 2 * dp] = pk2bf(g0.y, g1.y);
            *(unsigned*)&lBg[(fg + 2) * LDF + 2 * dp] = pk2bf(g0.z, g1.z);
            *(unsigned*)&lBg[(fg + 3) * LDF + 2 * dp] = pk2bf(g0.w, g1.w);
            *(unsigned*)&lBu[(fg + 0) * LDF + 2 * dp] = pk2bf(u0.x, u1.x);
            *(unsigned*)&lBu[(fg + 1) * LDF + 2 * dp] = pk2bf(u0.y, u1.y);
            *(unsigned*)&lBu[(fg + 2) * LDF + 2 * dp] = pk2bf(u0.z, u1.z);
            *(unsigned*)&lBu[(fg + 3) * LDF + 2 * dp] = pk2bf(u0.w, u1.w);
        }
        __syncthreads();

        FragAB a;
        const unsigned short* ap = &lA[(wm * 16 + lr) * LDF + half * 16];
        a.q[0] = *(const uint4*)ap;
        a.q[1] = *(const uint4*)(ap + 8);
#pragma unroll
        for (int ns = 0; ns < 4; ++ns) {
            const int n = ns * 16 + lr;
            FragAB bg, bu;
            const unsigned short* bgp = &lBg[n * LDF + half * 16];
            bg.q[0] = *(const uint4*)bgp;
            bg.q[1] = *(const uint4*)(bgp + 8);
            cg[ns] = __builtin_amdgcn_wmma_f32_16x16x32_bf16(false, a.v, false, bg.v,
                                                             (short)0, cg[ns], false, false);
            const unsigned short* bup = &lBu[n * LDF + half * 16];
            bu.q[0] = *(const uint4*)bup;
            bu.q[1] = *(const uint4*)(bup + 8);
            cu[ns] = __builtin_amdgcn_wmma_f32_16x16x32_bf16(false, a.v, false, bu.v,
                                                             (short)0, cu[ns], false, false);
        }
        __syncthreads();
    }

    // epilogue: h = silu(g)*u -> hbuf[slot][f]  (bf16), fast rcp for silu
#pragma unroll
    for (int ns = 0; ns < 4; ++ns) {
        const int f = f0 + ns * 16 + lr;
#pragma unroll
        for (int r = 0; r < 8; ++r) {
            const int m = wm * 16 + half * 8 + r;
            if (mtile * 128 + m < ce) {
                const int p = toks[m];
                float g = cg[ns][r], u = cu[ns][r];
                float s = __builtin_amdgcn_rcpf(1.f + __expf(-g));
                hbuf[(size_t)p * FDIM + f] = f2bf(g * s * u);
            }
        }
    }
}

// ---------------- kernel 6: down GEMM (bf16 WMMA), scaled by routing weight ----------------
__global__ __launch_bounds__(256) void k_gemm2(const unsigned short* __restrict__ hbuf,
                                               const float* __restrict__ Wd,
                                               const int* __restrict__ tix,
                                               const int* __restrict__ cnt,
                                               const float* __restrict__ wsel,
                                               float* __restrict__ ybuf) {
    const int e = blockIdx.z;
    const int mtile = blockIdx.y;
    const int d0 = blockIdx.x * 64;
    const int ce = cnt[e];
    if (mtile * 128 >= ce) return;

    __shared__ int   toks[128];
    __shared__ float wrow[128];
    __shared__ unsigned short lA[128 * LDF];
    __shared__ unsigned short lB[64 * LDF];

    const int tid = threadIdx.x;
    if (tid < 128) {
        int idx = mtile * 128 + tid;
        int p = (idx < ce) ? tix[e * T_TOK + idx] : 0;
        toks[tid] = p;
        wrow[tid] = wsel[p];
    }
    __syncthreads();

    const int wave = tid >> 5, lane = tid & 31;
    const int wm = wave;
    const int half = lane >> 4, lr = lane & 15;

    const int ar0 = tid >> 2, ac4 = tid & 3;
    const size_t abase0 = (size_t)toks[ar0] * FDIM + (size_t)ac4 * 8;
    const size_t abase1 = (size_t)toks[ar0 + 64] * FDIM + (size_t)ac4 * 8;

    const int dp = tid >> 4;
    const int fg = (tid & 15) * 4;        // here: d-column group

    const float* wdp = Wd + (size_t)e * FDIM * DDIM + d0;

    v8f zero = {0.f, 0.f, 0.f, 0.f, 0.f, 0.f, 0.f, 0.f};
    v8f cd[4];
#pragma unroll
    for (int ns = 0; ns < 4; ++ns) cd[ns] = zero;

    for (int k0 = 0; k0 < FDIM; k0 += 32) {
        *(uint4*)&lA[ar0 * LDF + ac4 * 8]        = *(const uint4*)(hbuf + abase0 + k0);
        *(uint4*)&lA[(ar0 + 64) * LDF + ac4 * 8] = *(const uint4*)(hbuf + abase1 + k0);
        {
            const float4 w0 = *(const float4*)(wdp + (size_t)(k0 + 2 * dp + 0) * DDIM + fg);
            const float4 w1 = *(const float4*)(wdp + (size_t)(k0 + 2 * dp + 1) * DDIM + fg);
            *(unsigned*)&lB[(fg + 0) * LDF + 2 * dp] = pk2bf(w0.x, w1.x);
            *(unsigned*)&lB[(fg + 1) * LDF + 2 * dp] = pk2bf(w0.y, w1.y);
            *(unsigned*)&lB[(fg + 2) * LDF + 2 * dp] = pk2bf(w0.z, w1.z);
            *(unsigned*)&lB[(fg + 3) * LDF + 2 * dp] = pk2bf(w0.w, w1.w);
        }
        __syncthreads();

        FragAB a;
        const unsigned short* ap = &lA[(wm * 16 + lr) * LDF + half * 16];
        a.q[0] = *(const uint4*)ap;
        a.q[1] = *(const uint4*)(ap + 8);
#pragma unroll
        for (int ns = 0; ns < 4; ++ns) {
            const int n = ns * 16 + lr;
            FragAB b;
            const unsigned short* bp = &lB[n * LDF + half * 16];
            b.q[0] = *(const uint4*)bp;
            b.q[1] = *(const uint4*)(bp + 8);
            cd[ns] = __builtin_amdgcn_wmma_f32_16x16x32_bf16(false, a.v, false, b.v,
                                                             (short)0, cd[ns], false, false);
        }
        __syncthreads();
    }

#pragma unroll
    for (int ns = 0; ns < 4; ++ns) {
        const int dc = d0 + ns * 16 + lr;
#pragma unroll
        for (int r = 0; r < 8; ++r) {
            const int m = wm * 16 + half * 8 + r;
            if (mtile * 128 + m < ce) {
                const int p = toks[m];
                ybuf[(size_t)p * DDIM + dc] = cd[ns][r] * wrow[m];
            }
        }
    }
}

// ---------------- kernel 7: combine the two routed results per token ----------------
__global__ __launch_bounds__(256) void k_combine(const float* __restrict__ ybuf,
                                                 float* __restrict__ out) {
    size_t i = (size_t)blockIdx.x * 256 + threadIdx.x;   // over T*D/4 float4s
    size_t t = i >> 8;                                   // D/4 == 256
    size_t cg = i & 255;
    const float4* y = (const float4*)ybuf;
    float4 a = y[(2 * t) * 256 + cg];
    float4 b = y[(2 * t + 1) * 256 + cg];
    float4 o = {a.x + b.x, a.y + b.y, a.z + b.z, a.w + b.w};
    ((float4*)out)[i] = o;
}

// ---------------- host launcher ----------------
extern "C" void kernel_launch(void* const* d_in, const int* in_sizes, int n_in,
                              void* d_out, int out_size, void* d_ws, size_t ws_size,
                              hipStream_t stream) {
    (void)in_sizes; (void)n_in; (void)out_size; (void)ws_size;
    const float* x  = (const float*)d_in[0];
    const float* gw = (const float*)d_in[1];
    const float* Wg = (const float*)d_in[2];
    const float* Wu = (const float*)d_in[3];
    const float* Wd = (const float*)d_in[4];

    float* out    = (float*)d_out;                          // (T, D)
    float* logits = out + (size_t)T_TOK * DDIM;             // (T, E)

    // workspace carve (all 16B aligned)
    char* w = (char*)d_ws;
    unsigned short* xb   = (unsigned short*)w; w += (size_t)T_TOK * DDIM * 2;          // 8 MB
    unsigned short* hbuf = (unsigned short*)w; w += (size_t)TOPK * T_TOK * FDIM * 2;   // 32 MB
    float* ybuf          = (float*)w;          w += (size_t)TOPK * T_TOK * DDIM * 4;   // 32 MB
    int*   sel           = (int*)w;            w += (size_t)T_TOK * TOPK * 4;
    float* wsel          = (float*)w;          w += (size_t)T_TOK * TOPK * 4;
    int*   tix           = (int*)w;            w += (size_t)NEXP * T_TOK * 4;
    int*   cnt           = (int*)w;            w += 256;

    k_cvt_x  <<<(T_TOK * DDIM / 4) / 256, 256, 0, stream>>>(x, xb);
    k_router <<<T_TOK / 8, 256, 0, stream>>>(x, gw, logits, sel, wsel);
    k_zero_cnt<<<1, 64, 0, stream>>>(cnt);
    k_scatter<<<(T_TOK * TOPK) / 256, 256, 0, stream>>>(sel, cnt, tix);
    k_gemm1  <<<dim3(FDIM / 64, T_TOK / 128, NEXP), 256, 0, stream>>>(xb, Wg, Wu, tix, cnt, hbuf);
    k_gemm2  <<<dim3(DDIM / 64, T_TOK / 128, NEXP), 256, 0, stream>>>(hbuf, Wd, tix, cnt, wsel, ybuf);
    k_combine<<<(T_TOK * DDIM / 4) / 256, 256, 0, stream>>>(ybuf, out);
}
